// EquivariantAttentionLayer_87273735455478
// MI455X (gfx1250) — matrix operand order repaired and checked by reading, not compiled, not run on hardware
//
#include <hip/hip_runtime.h>

// ---------------------------------------------------------------------------
// Problem constants (from the reference)
// ---------------------------------------------------------------------------
constexpr int Nf  = 128;    // frames
constexpr int Pp  = 256;    // points
constexpr int Dd  = 256;    // input dim
constexpr int Hh  = 16;     // heads
constexpr int Mm  = 32;     // head dim
constexpr int OUTC = Hh * Mm;   // 512
constexpr int HID  = 2048;
constexpr int NP   = Nf * Pp;   // 32768 rows
constexpr size_t G = (size_t)NP * OUTC;   // elements per Q/K/V plane

// ---------------------------------------------------------------------------
// WMMA types & helper (V_WMMA_F32_16X16X32_F16, wave32)
// ---------------------------------------------------------------------------
typedef __attribute__((ext_vector_type(16))) _Float16 v16h;
typedef __attribute__((ext_vector_type(8)))  float    v8f;
typedef __attribute__((ext_vector_type(4)))  int      v4i;

union F16x16 { v16h v; _Float16 h[16]; };
union F32x8  { v8f  v; float    f[8];  };

__device__ __forceinline__ v8f wmma_f16(const v16h& a, const v16h& b, const v8f& c) {
  // 8 args: (neg_a, A, neg_b, B, c_mod, C, reuse_a, reuse_b)
  return __builtin_amdgcn_wmma_f32_16x16x32_f16(false, a, false, b, (short)0, c,
                                                false, false);
}

// Lane maps used here, wave32 (A/B in contiguous k order, not the ISA table order):
//  A 16x32 f16: lane<16 -> kbBase=0, lane>=16 -> kbBase=16; element e -> K = kbBase + e
//  B 32x16 f16: lane<16 -> kbBase=0, lane>=16 -> kbBase=16; element e -> K = kbBase + e ; col = lane&15
//  C/D f32: vgpr r -> row = r + (lane<16 ? 0 : 8), col = lane&15

// ---------------------------------------------------------------------------
// Async global->LDS staging (GLOBAL_LOAD_ASYNC_TO_LDS_B128, ASYNCcnt).
// Builtin signature (from the round-2 diagnostic):
//   (v4i addrspace(1)* src, v4i addrspace(3)* dst, imm int offset, imm int cpol)
// Guarded: falls back to plain vector loads + ds stores if absent.
// ---------------------------------------------------------------------------
#ifndef __has_builtin
#define __has_builtin(x) 0
#endif

#if __has_builtin(__builtin_amdgcn_global_load_async_to_lds_b128)
#define USE_ASYNC_LDS 1
#else
#define USE_ASYNC_LDS 0
#endif

#if USE_ASYNC_LDS
typedef __attribute__((address_space(1))) v4i gv4i;   // global int4
typedef __attribute__((address_space(3))) v4i lv4i;   // LDS int4

__device__ __forceinline__ void async_copy16(const float* g, float* l) {
  __builtin_amdgcn_global_load_async_to_lds_b128(
      (gv4i*)g, (lv4i*)l, 0 /*imm offset*/, 0 /*cpol*/);
}
template <int N>
__device__ __forceinline__ void wait_async() {
#if __has_builtin(__builtin_amdgcn_s_wait_asynccnt)
  __builtin_amdgcn_s_wait_asynccnt((short)N);
#else
  asm volatile("s_wait_asynccnt %0" ::"n"(N) : "memory");
#endif
}
#else
__device__ __forceinline__ void async_copy16(const float* g, float* l) {
  *(float4*)l = *(const float4*)g;   // synchronous fallback
}
template <int N>
__device__ __forceinline__ void wait_async() {}
#endif

// ---------------------------------------------------------------------------
// Generic GEMM:  C(M_,N_) = A(M_,K_) @ B(K_,N_)  [+ bias]
//   BT=true : B is stored transposed, i.e. element (k,c) lives at B[c*K_ + k]
// Block = 128 threads (4 waves), block tile 64x64, wave tile 32x32, K step 32.
// Double-buffered LDS tiles filled by async B128 copies (16B per lane, 8 ops
// per K-step per wave); f32 in LDS, converted to f16 at fragment build.
// Row strides are multiples of 16B; stride 44 f32 gives conflict-free frag reads.
// ---------------------------------------------------------------------------
template <bool BT, bool BIAS>
__global__ __launch_bounds__(128)
void gemm_wmma(const float* __restrict__ A, const float* __restrict__ B,
               const float* __restrict__ bias, float* __restrict__ C,
               int M_, int N_, int K_) {
  constexpr int BRr = BT ? 64 : 32;   // B tile rows in LDS
  constexpr int BCc = BT ? 44 : 72;   // B tile row stride (f32)
  __shared__ alignas(16) float As[2][64][44];
  __shared__ alignas(16) float Bs[2][BRr][BCc];

  const int t    = threadIdx.x;
  const int lane = t & 31;
  const int wid  = t >> 5;
  const int br = blockIdx.y * 64;
  const int bc = blockIdx.x * 64;
  const int wr = (wid >> 1) * 32;
  const int wc = (wid & 1)  * 32;
  const int lrow   = lane & 15;
  const int kaBase = (lane < 16) ? 0 : 8;
  const int kbBase = (lane < 16) ? 0 : 16;
  const int rAdd   = (lane < 16) ? 0 : 8;

  // Stage one 64x32 A tile + 32x64 B tile into buffer `buf`
  // (8 async B128 ops per thread-group pass: 4 for A, 4 for B).
  auto stage = [&](int buf, int k0) {
#pragma unroll
    for (int q = 0; q < 4; ++q) {
      const int f = q * 512 + t * 4;            // A: 64*32 f32, 4 floats/lane/op
      const int r = f >> 5, k = f & 31;
      async_copy16(&A[(size_t)(br + r) * K_ + (k0 + k)], &As[buf][r][k]);
    }
#pragma unroll
    for (int q = 0; q < 4; ++q) {
      const int f = q * 512 + t * 4;
      if (BT) {                                  // B stored (col, k): copy along k
        const int c = f >> 5, k = f & 31;
        async_copy16(&B[(size_t)(bc + c) * K_ + (k0 + k)], &Bs[buf][c][k]);
      } else {                                   // B stored (k, col): copy along col
        const int k = f >> 6, c = f & 63;
        async_copy16(&B[(size_t)(k0 + k) * N_ + (bc + c)], &Bs[buf][k][c]);
      }
    }
  };

  v8f c00 = {}, c01 = {}, c10 = {}, c11 = {};

  const int nk = K_ / 32;
  stage(0, 0);
  for (int kt = 0; kt < nk; ++kt) {
    const int cur = kt & 1;
    if (kt + 1 < nk) {
      stage(cur ^ 1, (kt + 1) * 32);   // prefetch next K-step under the WMMAs
      wait_async<8>();                 // current buffer's 8 ops (oldest) done
    } else {
      wait_async<0>();
    }
    __syncthreads();

    F16x16 a0, a1, b0, b1;
#pragma unroll
    for (int e = 0; e < 16; ++e) {
      const int ka = kbBase + e;
      a0.h[e] = (_Float16)As[cur][wr + lrow][ka];
      a1.h[e] = (_Float16)As[cur][wr + 16 + lrow][ka];
      const int kb = kbBase + e;
      b0.h[e] = (_Float16)(BT ? Bs[cur][wc + lrow][kb]      : Bs[cur][kb][wc + lrow]);
      b1.h[e] = (_Float16)(BT ? Bs[cur][wc + 16 + lrow][kb] : Bs[cur][kb][wc + 16 + lrow]);
    }
    c00 = wmma_f16(a0.v, b0.v, c00);
    c01 = wmma_f16(a0.v, b1.v, c01);
    c10 = wmma_f16(a1.v, b0.v, c10);
    c11 = wmma_f16(a1.v, b1.v, c11);
    __syncthreads();   // all waves done with `cur` before it is restaged
  }

  F32x8 r00, r01, r10, r11;
  r00.v = c00; r01.v = c01; r10.v = c10; r11.v = c11;
  const int ccol = lane & 15;
#pragma unroll
  for (int r = 0; r < 8; ++r) {
    const int row0 = br + wr + r + rAdd;
    const int row1 = row0 + 16;
    const int col0 = bc + wc + ccol;
    const int col1 = col0 + 16;
    const float add0 = BIAS ? bias[col0] : 0.0f;
    const float add1 = BIAS ? bias[col1] : 0.0f;
    C[(size_t)row0 * N_ + col0] = r00.f[r] + add0;
    C[(size_t)row0 * N_ + col1] = r01.f[r] + add1;
    C[(size_t)row1 * N_ + col0] = r10.f[r] + add0;
    C[(size_t)row1 * N_ + col1] = r11.f[r] + add1;
  }
}

// ---------------------------------------------------------------------------
// Temporal attention: one block per (point j, head h).
// Q,K,V: 128 frames x 32.  S = Q K^T (128x128), row softmax over key frame,
// then T = A^T V  (reproduces the reference's 'hIij' transposed-weights einsum).
// ---------------------------------------------------------------------------
__global__ __launch_bounds__(128)
void temporal_attn(const float* __restrict__ QKVt, float* __restrict__ outT) {
  __shared__ _Float16 Qs[128][32];
  __shared__ _Float16 Ks[128][32];
  __shared__ _Float16 Vs[128][32];
  __shared__ float    Sc[64][128];    // one 64-row score chunk (f32)
  __shared__ _Float16 Aw[128][128];   // normalized weights (f16)

  const int j = blockIdx.x;
  const int h = blockIdx.y;
  const int t = threadIdx.x;
  const int lane = t & 31, wid = t >> 5;
  const int lrow   = lane & 15;
  const int kaBase = (lane < 16) ? 0 : 8;
  const int kbBase = (lane < 16) ? 0 : 16;
  const int rAdd   = (lane < 16) ? 0 : 8;

  for (int e = t; e < 128 * 32; e += 128) {
    const int i = e >> 5, m = e & 31;
    const size_t base = ((size_t)i * Pp + j) * OUTC + h * Mm + m;
    Qs[i][m] = (_Float16)QKVt[base];
    Ks[i][m] = (_Float16)QKVt[G + base];
    Vs[i][m] = (_Float16)QKVt[2 * G + base];
  }
  __syncthreads();

  // scores + softmax, two 64-row chunks; wave wid owns row-tile wid of chunk
  for (int c = 0; c < 2; ++c) {
    const int i0 = c * 64;
    F16x16 a;
#pragma unroll
    for (int e = 0; e < 16; ++e)
      a.h[e] = Qs[i0 + wid * 16 + lrow][kbBase + e];
    for (int ct = 0; ct < 8; ++ct) {
      F16x16 b;
#pragma unroll
      for (int e = 0; e < 16; ++e)
        b.h[e] = Ks[ct * 16 + lrow][kbBase + e];   // B = K^T
      v8f acc = {};
      acc = wmma_f16(a.v, b.v, acc);
      F32x8 rr; rr.v = acc;
#pragma unroll
      for (int r = 0; r < 8; ++r)
        Sc[wid * 16 + r + rAdd][ct * 16 + lrow] = rr.f[r];
    }
    __syncthreads();
    if (t < 64) {
      float mx = -1e30f;
      for (int I = 0; I < 128; ++I) mx = fmaxf(mx, Sc[t][I]);
      float sum = 0.0f;
      for (int I = 0; I < 128; ++I) sum += __expf(Sc[t][I] - mx);
      const float inv = 1.0f / sum;
      for (int I = 0; I < 128; ++I)
        Aw[i0 + t][I] = (_Float16)(__expf(Sc[t][I] - mx) * inv);
    }
    __syncthreads();
  }

  // T = A^T @ V  (output 128 x 32). Wave wid -> row-tiles {2wid, 2wid+1}.
#pragma unroll
  for (int rsub = 0; rsub < 2; ++rsub) {
    const int rt = wid * 2 + rsub;
#pragma unroll
    for (int ctl = 0; ctl < 2; ++ctl) {
      v8f acc = {};
      for (int kt = 0; kt < 4; ++kt) {
        F16x16 a, b;
#pragma unroll
        for (int e = 0; e < 16; ++e) {
          a.h[e] = Aw[kt * 32 + kbBase + e][rt * 16 + lrow]; // A^T read
          b.h[e] = Vs[kt * 32 + kbBase + e][ctl * 16 + lrow];
        }
        acc = wmma_f16(a.v, b.v, acc);
      }
      F32x8 rr; rr.v = acc;
#pragma unroll
      for (int r = 0; r < 8; ++r) {
        const int i = rt * 16 + r + rAdd;
        const int m = ctl * 16 + lrow;
        outT[((size_t)i * Pp + j) * OUTC + h * Mm + m] = rr.f[r];
      }
    }
  }
}

// ---------------------------------------------------------------------------
// Point attention: one block per (frame i, head h).
// Q,K,V: 256 points x 32. Processed in four 64-row chunks:
//   S chunk (64x256) -> row softmax -> pa chunk = A @ V (64x32).
// ---------------------------------------------------------------------------
__global__ __launch_bounds__(128)
void point_attn(const float* __restrict__ QKVp, float* __restrict__ outPA) {
  __shared__ _Float16 Qs[256][32];
  __shared__ _Float16 Ks[256][32];
  __shared__ _Float16 Vs[256][32];
  __shared__ float    Sc[64][256];
  __shared__ _Float16 Ac[64][256];

  const int i = blockIdx.x;
  const int h = blockIdx.y;
  const int t = threadIdx.x;
  const int lane = t & 31, wid = t >> 5;
  const int lrow   = lane & 15;
  const int kaBase = (lane < 16) ? 0 : 8;
  const int kbBase = (lane < 16) ? 0 : 16;
  const int rAdd   = (lane < 16) ? 0 : 8;

  for (int e = t; e < 256 * 32; e += 128) {
    const int jj = e >> 5, m = e & 31;
    const size_t base = ((size_t)i * Pp + jj) * OUTC + h * Mm + m;
    Qs[jj][m] = (_Float16)QKVp[base];
    Ks[jj][m] = (_Float16)QKVp[G + base];
    Vs[jj][m] = (_Float16)QKVp[2 * G + base];
  }
  __syncthreads();

  for (int c = 0; c < 4; ++c) {
    const int r0 = c * 64;
    F16x16 a;
#pragma unroll
    for (int e = 0; e < 16; ++e)
      a.h[e] = Qs[r0 + wid * 16 + lrow][kbBase + e];
    for (int ct = 0; ct < 16; ++ct) {
      F16x16 b;
#pragma unroll
      for (int e = 0; e < 16; ++e)
        b.h[e] = Ks[ct * 16 + lrow][kbBase + e];
      v8f acc = {};
      acc = wmma_f16(a.v, b.v, acc);
      F32x8 rr; rr.v = acc;
#pragma unroll
      for (int r = 0; r < 8; ++r)
        Sc[wid * 16 + r + rAdd][ct * 16 + lrow] = rr.f[r];
    }
    __syncthreads();
    if (t < 64) {
      float mx = -1e30f;
      for (int J = 0; J < 256; ++J) mx = fmaxf(mx, Sc[t][J]);
      float sum = 0.0f;
      for (int J = 0; J < 256; ++J) sum += __expf(Sc[t][J] - mx);
      const float inv = 1.0f / sum;
      for (int J = 0; J < 256; ++J)
        Ac[t][J] = (_Float16)(__expf(Sc[t][J] - mx) * inv);
    }
    __syncthreads();

    // pa chunk = A @ V ; wave wid -> row-tile wid, col-tiles {0,1}
#pragma unroll
    for (int ctl = 0; ctl < 2; ++ctl) {
      v8f acc = {};
      for (int kt = 0; kt < 8; ++kt) {
        F16x16 a2, b2;
#pragma unroll
        for (int e = 0; e < 16; ++e) {
          a2.h[e] = Ac[wid * 16 + lrow][kt * 32 + kbBase + e];
          b2.h[e] = Vs[kt * 32 + kbBase + e][ctl * 16 + lrow];
        }
        acc = wmma_f16(a2.v, b2.v, acc);
      }
      F32x8 rr; rr.v = acc;
#pragma unroll
      for (int r = 0; r < 8; ++r) {
        const int jj = r0 + wid * 16 + r + rAdd;
        const int m  = ctl * 16 + lrow;
        outPA[((size_t)i * Pp + jj) * OUTC + h * Mm + m] = rr.f[r];
      }
    }
    __syncthreads();   // before next chunk reuses Sc/Ac
  }
}

// ---------------------------------------------------------------------------
// Launch: 6 stages, buffers reused inside d_ws (needs 512 MB):
//   QKV plane buf (192MB) | temporal/pa buf (64MB) | MLP hidden (256MB)
// ---------------------------------------------------------------------------
extern "C" void kernel_launch(void* const* d_in, const int* in_sizes, int n_in,
                              void* d_out, int out_size, void* d_ws, size_t ws_size,
                              hipStream_t stream) {
  (void)in_sizes; (void)n_in; (void)out_size; (void)ws_size;
  const float* x  = (const float*)d_in[0];
  const float* Wt = (const float*)d_in[1];   // (3, D, H, M) -> per-w (256 x 512) row-major
  const float* Wp = (const float*)d_in[2];   // (3, H, M, H, M) -> per-w stored (out=512, in=512): transposed B
  const float* W1 = (const float*)d_in[3];   // (512, 2048)
  const float* b1 = (const float*)d_in[4];
  const float* W2 = (const float*)d_in[5];   // (2048, 512)
  const float* b2 = (const float*)d_in[6];
  float* out = (float*)d_out;

  float* ws  = (float*)d_ws;
  float* QKV = ws;                 // 3 * NP * OUTC floats
  float* Tmp = ws + 3 * G;         // NP * OUTC floats (temporal, then pa)
  float* Hid = ws + 4 * G;         // NP * HID floats

  const dim3 blk(128);

  // 1) temporal QKV: (NP x D) @ (D x 512) for w=0..2
  for (int w = 0; w < 3; ++w)
    gemm_wmma<false, false><<<dim3(OUTC / 64, NP / 64), blk, 0, stream>>>(
        x, Wt + (size_t)w * Dd * OUTC, nullptr, QKV + (size_t)w * G, NP, OUTC, Dd);

  // 2) temporal attention over frames (with the reference's A^T quirk)
  temporal_attn<<<dim3(Pp, Hh), blk, 0, stream>>>(QKV, Tmp);

  // 3) point QKV: (NP x 512) @ (512 x 512), B stored transposed
  for (int w = 0; w < 3; ++w)
    gemm_wmma<true, false><<<dim3(OUTC / 64, NP / 64), blk, 0, stream>>>(
        Tmp, Wp + (size_t)w * OUTC * OUTC, nullptr, QKV + (size_t)w * G, NP, OUTC, OUTC);

  // 4) point attention over points
  point_attn<<<dim3(Nf, Hh), blk, 0, stream>>>(QKV, Tmp);

  // 5) MLP layer 1: (NP x 512) @ (512 x 2048) + b1
  gemm_wmma<false, true><<<dim3(HID / 64, NP / 64), blk, 0, stream>>>(
      Tmp, W1, b1, Hid, NP, HID, OUTC);

  // 6) MLP layer 2: (NP x 2048) @ (2048 x 512) + b2 -> output
  gemm_wmma<false, true><<<dim3(OUTC / 64, NP / 64), blk, 0, stream>>>(
      Hid, W2, b2, out, NP, OUTC, HID);
}
